// RPN_59064390255256
// MI455X (gfx1250) — matrix operand order repaired and checked by reading, not compiled, not run on hardware
//
#include <hip/hip_runtime.h>
#include <hip/hip_bf16.h>
#include <math.h>

typedef __attribute__((ext_vector_type(16))) _Float16 v16h;
typedef __attribute__((ext_vector_type(8)))  float    v8f;

#define NUM_LEVELS 5
#define TOTAL_ANCH 268569     // sum over levels of H*W*3
#define PRE_NMS    6000
#define POST_NMS   1000
#define PAD_SORT   (1 << 19)  // 524288 >= 268569
#define NBATCH     2

union H8 { _Float16 h[8]; uint4 v; };

// ---------------------------------------------------------------------------
// Fused conv3x3(256->256)+ReLU + 1x1 heads + sigmoid + box decode/clip.
// One block = 64 output positions of one level / one image.
// GEMM: C[256 x 64] = Wconv[256 x 2304] * Im2col[2304 x 64] via WMMA f16,
// f32 accumulate.  LDS tiles stored in WMMA fragment order so each fragment
// is two ds_load_b128; K-chunks double-buffered (1 barrier / chunk).
// ---------------------------------------------------------------------------
__global__ void __launch_bounds__(256)
rpn_fused_level(const float* __restrict__ feat,      // [N,256,H,W]
                const float* __restrict__ convw,     // [256,2304]
                const float* __restrict__ convb,     // [256]
                const float* __restrict__ clsw,      // [3,256]
                const float* __restrict__ clsb,      // [3]
                const float* __restrict__ bboxw,     // [12,256]
                const float* __restrict__ bboxb,     // [12]
                const float* __restrict__ im_info,   // [N,6]
                float* __restrict__ scoresAll,       // [N,TOTAL_ANCH]
                float* __restrict__ boxesAll,        // [N,TOTAL_ANCH,4]
                int H, int W, int stride, float scale, int lvloff)
{
    constexpr int KTOT = 2304, KC = 32, TP = 64, NCHUNK = KTOT / KC;
    // A tile rows are K-permuted {0..7,16..23,8..15,24..31}: a lane-half's
    // 16 fragment halfs are contiguous.  B tile is [p][k].
    __shared__ __attribute__((aligned(32))) _Float16 Ash[2][256 * KC]; // 2x16KB
    __shared__ __attribute__((aligned(32))) _Float16 Bsh[2][TP * KC];  // 2x 4KB
    __shared__ float headw[15 * 256]; // 3 cls + 12 bbox 1x1 weights
    __shared__ float convb_sh[256];
    __shared__ float headacc[TP * 15];

    const int tid  = threadIdx.x;
    const int wv   = tid >> 5;          // wave id 0..7
    const int lane = tid & 31;
    const int n    = blockIdx.y;
    const int tile0 = blockIdx.x * TP;
    const int HW = H * W;

    for (int i = tid; i < 15 * 256; i += 256) {
        int j = i >> 8, co = i & 255;
        headw[i] = (j < 3) ? clsw[j * 256 + co] : bboxw[(j - 3) * 256 + co];
    }
    convb_sh[tid] = convb[tid];
    for (int i = tid; i < TP * 15; i += 256) headacc[i] = 0.f;

    v8f acc[2][4];
#pragma unroll
    for (int mt = 0; mt < 2; ++mt)
#pragma unroll
        for (int nt = 0; nt < 4; ++nt)
#pragma unroll
            for (int v = 0; v < 8; ++v) acc[mt][nt][v] = 0.f;

    const float* featn = feat + (size_t)n * 256 * HW;
    const int mr = lane & 15, hf = lane >> 4, col = lane & 15;

    // stage K-chunk `kc` into buffer `buf` -------------------------------
    auto stage = [&](int kc, int buf) {
        const int kbase = kc * KC;
        // A: 256x32 halfs, permuted runs; 4 x uint4 (8 halfs) per thread.
        uint4* A4 = (uint4*)&Ash[buf][0];
#pragma unroll
        for (int t = 0; t < 4; ++t) {
            int i4  = tid + t * 256;          // 1024 x 16B
            int co  = i4 >> 2;
            int run = i4 & 3;                 // slot-run -> K-run base
            int kk0 = (run == 0) ? 0 : (run == 1) ? 16 : (run == 2) ? 8 : 24;
            const float* src = convw + (size_t)co * KTOT + kbase + kk0;
            H8 pk;
#pragma unroll
            for (int q = 0; q < 8; ++q) pk.h[q] = (_Float16)src[q];
            A4[i4] = pk.v;
        }
        // B: 64 positions x 32 halfs ([p][k]); 1 x uint4 per thread.
        uint4* B4 = (uint4*)&Bsh[buf][0];
        {
            int i4  = tid;                    // 256 x 16B
            int p   = i4 >> 2;
            int kk0 = (((i4 & 3) & 1) << 4) | (((i4 & 3) & 2) << 2);
            int pos = tile0 + p;
            int y = 0, x = 0;
            bool pin = pos < HW;
            if (pin) { y = pos / W; x = pos - y * W; }
            H8 pk;
#pragma unroll
            for (int q = 0; q < 8; ++q) {
                int k  = kbase + kk0 + q;
                int ci = k / 9;
                int r  = k - ci * 9;
                int ky = r / 3;
                int kx = r - ky * 3;
                float v = 0.f;
                if (pin) {
                    int iy = y + ky - 1, ix = x + kx - 1;
                    if (iy >= 0 && iy < H && ix >= 0 && ix < W)
                        v = featn[(size_t)ci * HW + iy * W + ix];
                }
                pk.h[q] = (_Float16)v;
            }
            B4[i4] = pk.v;
        }
    };

    stage(0, 0);
    for (int kc = 0; kc < NCHUNK; ++kc) {
        __syncthreads();
        const int cb = kc & 1;
        if (kc + 1 < NCHUNK) stage(kc + 1, cb ^ 1);

        // fragments: contiguous 32B per lane -> 2x ds_load_b128 each
        const v16h a0 = *(const v16h*)&Ash[cb][(wv * 32 + mr) * KC + hf * 16];
        const v16h a1 = *(const v16h*)&Ash[cb][(wv * 32 + 16 + mr) * KC + hf * 16];
#pragma unroll
        for (int nt = 0; nt < 4; ++nt) {
            const v16h b = *(const v16h*)&Bsh[cb][(nt * 16 + col) * KC + hf * 16];
            acc[0][nt] = __builtin_amdgcn_wmma_f32_16x16x32_f16(
                false, a0, false, b, (short)0, acc[0][nt], false, false);
            acc[1][nt] = __builtin_amdgcn_wmma_f32_16x16x32_f16(
                false, a1, false, b, (short)0, acc[1][nt], false, false);
        }
    }
    __syncthreads();

    // bias + ReLU + 1x1 heads (cls 3, bbox 12); reduce across waves via LDS.
#pragma unroll
    for (int nt = 0; nt < 4; ++nt) {
        float ph[15];
#pragma unroll
        for (int j = 0; j < 15; ++j) ph[j] = 0.f;
#pragma unroll
        for (int mt = 0; mt < 2; ++mt)
#pragma unroll
            for (int v = 0; v < 8; ++v) {
                int co = wv * 32 + mt * 16 + hf * 8 + v; // C layout: VGPR v -> row hf*8+v
                float tv = acc[mt][nt][v] + convb_sh[co];
                tv = fmaxf(tv, 0.f);
#pragma unroll
                for (int j = 0; j < 15; ++j) ph[j] += tv * headw[j * 256 + co];
            }
        int pl = nt * 16 + col;
#pragma unroll
        for (int j = 0; j < 15; ++j) atomicAdd(&headacc[pl * 15 + j], ph[j]);
    }
    __syncthreads();

    // sigmoid + anchor decode + clip, write scores & boxes
    const float imh = im_info[n * 6 + 0];
    const float imw = im_info[n * 6 + 1];
    for (int e = tid; e < TP * 3; e += 256) {
        int pl = e / 3, a = e - pl * 3;
        int pos = tile0 + pl;
        if (pos >= HW) continue;
        size_t gi = (size_t)n * TOTAL_ANCH + lvloff + (size_t)pos * 3 + a;

        float sv = headacc[pl * 15 + a] + clsb[a];
        scoresAll[gi] = 1.f / (1.f + expf(-sv));

        float dx = headacc[pl * 15 + 3 + a * 4 + 0] + bboxb[a * 4 + 0];
        float dy = headacc[pl * 15 + 3 + a * 4 + 1] + bboxb[a * 4 + 1];
        float dw = headacc[pl * 15 + 3 + a * 4 + 2] + bboxb[a * 4 + 2];
        float dh = headacc[pl * 15 + 3 + a * 4 + 3] + bboxb[a * 4 + 3];

        // base anchor (BASE_SIZE=16, ratios 1,2,3)
        float ratio = (float)(a + 1);
        float w0 = roundf(sqrtf(256.f / ratio));
        float h0 = roundf(w0 * ratio);
        float aws = w0 * scale, ahs = h0 * scale;
        int   y = pos / W, x = pos - y * W;
        float sx = (float)(x * stride), sy = (float)(y * stride);
        float ax1 = 7.5f - 0.5f * (aws - 1.f) + sx;
        float ay1 = 7.5f - 0.5f * (ahs - 1.f) + sy;
        float ax2 = 7.5f + 0.5f * (aws - 1.f) + sx;
        float ay2 = 7.5f + 0.5f * (ahs - 1.f) + sy;

        float aw = ax2 - ax1 + 1.f, ah = ay2 - ay1 + 1.f;
        float acx = ax1 + 0.5f * (aw - 1.f), acy = ay1 + 0.5f * (ah - 1.f);
        float cx = dx * aw + acx, cy = dy * ah + acy;
        float bw = expf(dw) * aw, bh = expf(dh) * ah;
        float x1 = cx - 0.5f * (bw - 1.f), y1 = cy - 0.5f * (bh - 1.f);
        float x2 = cx + 0.5f * (bw - 1.f), y2 = cy + 0.5f * (bh - 1.f);
        x1 = fminf(fmaxf(x1, 0.f), imw - 1.f);
        y1 = fminf(fmaxf(y1, 0.f), imh - 1.f);
        x2 = fminf(fmaxf(x2, 0.f), imw - 1.f);
        y2 = fminf(fmaxf(y2, 0.f), imh - 1.f);
        float* bo = boxesAll + gi * 4;
        bo[0] = x1; bo[1] = y1; bo[2] = x2; bo[3] = y2;
    }
}

// ---------------------------------------------------------------------------
// Top-k via padded bitonic sort (descending), per batch image.
// ---------------------------------------------------------------------------
__global__ void sort_init(const float* __restrict__ scoresAll,
                          float* __restrict__ keys, int* __restrict__ sidx)
{
    int i = blockIdx.x * blockDim.x + threadIdx.x;
    int n = blockIdx.y;
    if (i >= PAD_SORT) return;
    size_t o = (size_t)n * PAD_SORT + i;
    keys[o] = (i < TOTAL_ANCH) ? scoresAll[(size_t)n * TOTAL_ANCH + i]
                               : -__builtin_inff();
    sidx[o] = i;
}

__global__ void bitonic_step(float* __restrict__ keys, int* __restrict__ sidx,
                             int k, int j)
{
    int i = blockIdx.x * blockDim.x + threadIdx.x;
    if (i >= PAD_SORT) return;
    int n = blockIdx.y;
    float* kk = keys + (size_t)n * PAD_SORT;
    int*   ii = sidx + (size_t)n * PAD_SORT;
    int ixj = i ^ j;
    if (ixj > i) {
        bool up = (i & k) == 0;               // 'up' segments -> descending
        float a = kk[i], b = kk[ixj];
        bool sw = up ? (a < b) : (a > b);
        if (sw) {
            kk[i] = b; kk[ixj] = a;
            int t = ii[i]; ii[i] = ii[ixj]; ii[ixj] = t;
        }
    }
}

// ---------------------------------------------------------------------------
// Greedy NMS over top-6000 (sorted desc) + ROI emit. One block per image.
// ---------------------------------------------------------------------------
__global__ void __launch_bounds__(256)
nms_emit(const int* __restrict__ sidx, const float* __restrict__ boxesAll,
         float* __restrict__ topb, float* __restrict__ out_rois,
         int* __restrict__ out_inds)
{
    __shared__ unsigned char sup[PRE_NMS];
    __shared__ int keep[POST_NMS];
    const int n = blockIdx.x, tid = threadIdx.x;
    const int* si = sidx + (size_t)n * PAD_SORT;
    float* tb = topb + (size_t)n * PRE_NMS * 4;

    for (int r = tid; r < PRE_NMS; r += blockDim.x) {
        int g = si[r];
        const float* b = boxesAll + ((size_t)n * TOTAL_ANCH + g) * 4;
        tb[r * 4 + 0] = b[0]; tb[r * 4 + 1] = b[1];
        tb[r * 4 + 2] = b[2]; tb[r * 4 + 3] = b[3];
        sup[r] = 0;
    }
    __syncthreads();

    for (int i = 0; i < PRE_NMS; ++i) {
        if (!sup[i]) {
            float ax1 = tb[i * 4 + 0], ay1 = tb[i * 4 + 1];
            float ax2 = tb[i * 4 + 2], ay2 = tb[i * 4 + 3];
            float areai = (ax2 - ax1 + 1.f) * (ay2 - ay1 + 1.f);
            for (int j = i + 1 + tid; j < PRE_NMS; j += blockDim.x) {
                if (sup[j]) continue;
                float bx1 = tb[j * 4 + 0], by1 = tb[j * 4 + 1];
                float bx2 = tb[j * 4 + 2], by2 = tb[j * 4 + 3];
                float ix1 = fmaxf(ax1, bx1), iy1 = fmaxf(ay1, by1);
                float ix2 = fminf(ax2, bx2), iy2 = fminf(ay2, by2);
                float iw = fmaxf(ix2 - ix1 + 1.f, 0.f);
                float ih = fmaxf(iy2 - iy1 + 1.f, 0.f);
                float inter = iw * ih;
                float areaj = (bx2 - bx1 + 1.f) * (by2 - by1 + 1.f);
                float iou = inter / (areai + areaj - inter);
                if (iou > 0.7f) sup[j] = 1;
            }
        }
        __syncthreads();
    }

    if (tid == 0) {   // ordered compaction == top_k over (-inf masked) scores
        int c = 0;
        for (int r = 0; r < PRE_NMS && c < POST_NMS; ++r)
            if (!sup[r]) keep[c++] = r;
        for (int r = 0; r < PRE_NMS && c < POST_NMS; ++r)
            if (sup[r]) keep[c++] = r;
    }
    __syncthreads();

    for (int r = tid; r < POST_NMS; r += blockDim.x) {
        int ridx = keep[r];
        float* o = out_rois + ((size_t)n * POST_NMS + r) * 5;
        o[0] = (float)n;
        o[1] = tb[ridx * 4 + 0]; o[2] = tb[ridx * 4 + 1];
        o[3] = tb[ridx * 4 + 2]; o[4] = tb[ridx * 4 + 3];
        out_inds[n * POST_NMS + r] = n;
    }
}

// ---------------------------------------------------------------------------
extern "C" void kernel_launch(void* const* d_in, const int* in_sizes, int n_in,
                              void* d_out, int out_size, void* d_ws, size_t ws_size,
                              hipStream_t stream)
{
    (void)in_sizes; (void)n_in; (void)out_size; (void)ws_size;
    const float* feats[5] = {(const float*)d_in[0], (const float*)d_in[1],
                             (const float*)d_in[2], (const float*)d_in[3],
                             (const float*)d_in[4]};
    const float* im_info = (const float*)d_in[5];
    const float* convw   = (const float*)d_in[6];
    const float* convb   = (const float*)d_in[7];
    const float* clsw    = (const float*)d_in[8];
    const float* clsb    = (const float*)d_in[9];
    const float* bboxw   = (const float*)d_in[10];
    const float* bboxb   = (const float*)d_in[11];

    // workspace carve-up (256B aligned)
    char* ws = (char*)d_ws;
    size_t off = 0;
    auto take = [&](size_t bytes) -> char* {
        char* p = ws + off;
        off = (off + bytes + 255) & ~(size_t)255;
        return p;
    };
    float* scoresAll = (float*)take((size_t)NBATCH * TOTAL_ANCH * 4);
    float* boxesAll  = (float*)take((size_t)NBATCH * TOTAL_ANCH * 16);
    float* keys      = (float*)take((size_t)NBATCH * PAD_SORT * 4);
    int*   sidx      = (int*)  take((size_t)NBATCH * PAD_SORT * 4);
    float* topb      = (float*)take((size_t)NBATCH * PRE_NMS * 16);

    static const int Hs[5]      = {13, 25, 50, 100, 200};
    static const int Ws_[5]     = {21, 42, 84, 168, 336};
    static const int strides[5] = {64, 32, 16, 8, 4};
    static const float scales[5] = {32.f, 16.f, 8.f, 4.f, 2.f};

    int lvloff = 0;
    for (int l = 0; l < NUM_LEVELS; ++l) {
        int HW = Hs[l] * Ws_[l];
        dim3 grid((HW + 63) / 64, NBATCH);
        rpn_fused_level<<<grid, 256, 0, stream>>>(
            feats[l], convw, convb, clsw, clsb, bboxw, bboxb, im_info,
            scoresAll, boxesAll, Hs[l], Ws_[l], strides[l], scales[l], lvloff);
        lvloff += HW * 3;
    }

    sort_init<<<dim3(PAD_SORT / 256, NBATCH), 256, 0, stream>>>(scoresAll, keys, sidx);
    for (int k = 2; k <= PAD_SORT; k <<= 1)
        for (int j = k >> 1; j > 0; j >>= 1)
            bitonic_step<<<dim3(PAD_SORT / 256, NBATCH), 256, 0, stream>>>(keys, sidx, k, j);

    float* out_rois = (float*)d_out;                         // [2*1000, 5]
    int*   out_inds = (int*)d_out + NBATCH * POST_NMS * 5;   // [2*1000] int32
    nms_emit<<<dim3(NBATCH), 256, 0, stream>>>(sidx, boxesAll, topb, out_rois, out_inds);
}
